// LongConv_65197603553740
// MI455X (gfx1250) — matrix operand order repaired and checked by reading, not compiled, not run on hardware
//
#include <hip/hip_runtime.h>
#include <hip/hip_bf16.h>

// ---------------------------------------------------------------------------
// LongConv (FlashFFTConv-style) for MI455X / gfx1250
//   B=4, H=1024, L=4096, C=1, n=2L=8192
//   Stage 1: soft-threshold kernel + FFT(8192)      -> kf   (ws)
//   Stage 2: W f32 -> f16                           -> Wf16 (ws)
//   Stage 3: FFT conv per (b,h) + D skip + SiLU     -> ybf  f16, (B*L) x H
//   Stage 4: WMMA GEMM  z = ybf * W^T  (TDM-fed)    -> z    f32, M=16384,N=2048,K=1024
//   Stage 5: GLU (LDS 32x32 transpose)              -> d_out (B,H,L)
// ---------------------------------------------------------------------------

#define FFT_N    8192
#define FFT_LOG2 13
#define LEN_L    4096
#define DIM_H    1024
#define BATCH_B  4
#define TWO_PI   6.28318530717958647692f

typedef __attribute__((ext_vector_type(16))) _Float16 v16h;
typedef __attribute__((ext_vector_type(8)))  _Float16 v8h;
typedef __attribute__((ext_vector_type(8)))  float    v8f;
typedef __attribute__((ext_vector_type(4)))  unsigned int u32x4;
typedef __attribute__((ext_vector_type(4)))  int          i32x4;
typedef __attribute__((ext_vector_type(8)))  int          i32x8;

#if defined(__has_builtin)
#if __has_builtin(__builtin_amdgcn_tensor_load_to_lds) && \
    __has_builtin(__builtin_amdgcn_s_wait_tensorcnt)
#define HAVE_TDM 1
#endif
#endif
#ifndef HAVE_TDM
#define HAVE_TDM 0
#endif

__device__ __forceinline__ float2 cmul(float2 a, float2 b) {
    return make_float2(a.x * b.x - a.y * b.y, a.x * b.y + a.y * b.x);
}

__device__ __forceinline__ int bitrev13(int i) {
    return (int)(__brev((unsigned)i) >> (32 - FFT_LOG2));
}

// In-place radix-2 DIT FFT over LDS; input must be in bit-reversed order.
__device__ void fft_inplace(float2* s, float sign, int tid, int nthr) {
    for (int lh = 0; lh < FFT_LOG2; ++lh) {
        const int half = 1 << lh;
        const float theta = sign * TWO_PI / (float)(half << 1);
        for (int t = tid; t < (FFT_N >> 1); t += nthr) {
            const int j    = t & (half - 1);
            const int base = (t >> lh) << (lh + 1);
            float sn, cs;
            __sincosf(theta * (float)j, &sn, &cs);
            const float2 w  = make_float2(cs, sn);
            const float2 a  = s[base + j];
            const float2 bb = s[base + j + half];
            const float2 tt = cmul(w, bb);
            s[base + j]        = make_float2(a.x + tt.x, a.y + tt.y);
            s[base + j + half] = make_float2(a.x - tt.x, a.y - tt.y);
        }
        __syncthreads();
    }
}

// ---------------------------------------------------------------------------
// Kernel 1: soft-threshold + forward FFT of conv kernel rows -> kf[h][0..8191]
// ---------------------------------------------------------------------------
__global__ void kfft_kernel(const float* __restrict__ kern,
                            float2* __restrict__ kf) {
    extern __shared__ float2 smem[];
    const int h   = blockIdx.x;
    const int tid = threadIdx.x;
    const int nth = blockDim.x;
    const float* krow = kern + (size_t)h * FFT_N;
    for (int i = tid; i < FFT_N; i += nth) {
        const float kv  = krow[i];
        const float mag = fabsf(kv) - 0.1f;           // KERNEL_LAM = 0.1
        const float v   = (mag > 0.0f) ? copysignf(mag, kv) : 0.0f;
        smem[bitrev13(i)] = make_float2(v, 0.0f);
    }
    __syncthreads();
    fft_inplace(smem, -1.0f, tid, nth);
    float2* orow = kf + (size_t)h * FFT_N;
    for (int i = tid; i < FFT_N; i += nth) orow[i] = smem[i];
}

// ---------------------------------------------------------------------------
// Kernel 2: W (f32, [2H][H] row-major = [n][k]) -> f16 (B matrix, col-major)
// ---------------------------------------------------------------------------
__global__ void wconv_kernel(const float* __restrict__ W,
                             _Float16* __restrict__ Wf) {
    const size_t i = (size_t)blockIdx.x * blockDim.x + threadIdx.x;
    Wf[i] = (_Float16)W[i];
}

// ---------------------------------------------------------------------------
// Kernel 3: per-(b,h) FFT convolution + D skip + SiLU -> f16 A-matrix
//   ybf[(b*L + l)*H + h]    (scattered 2B stores; ybf = 32MB, lives in L2)
// ---------------------------------------------------------------------------
__global__ void conv_kernel(const float* __restrict__ u,
                            const float* __restrict__ D,
                            const float2* __restrict__ kf,
                            _Float16* __restrict__ ybf) {
    extern __shared__ float2 smem[];
    const int blk = blockIdx.x;
    const int b   = blk >> 10;
    const int h   = blk & (DIM_H - 1);
    const int tid = threadIdx.x;
    const int nth = blockDim.x;

    const float* urow = u + ((size_t)b * DIM_H + h) * LEN_L;

    for (int i = tid; i < FFT_N; i += nth) {
        const float v = (i < LEN_L) ? urow[i] : 0.0f;
        smem[bitrev13(i)] = make_float2(v, 0.0f);
    }
    __syncthreads();
    fft_inplace(smem, -1.0f, tid, nth);

    const float2* krow = kf + (size_t)h * FFT_N;
    for (int i = tid; i < FFT_N; i += nth) smem[i] = cmul(smem[i], krow[i]);
    __syncthreads();

    for (int i = tid; i < FFT_N; i += nth) {
        const int r = bitrev13(i);
        if (r > i) { const float2 t = smem[i]; smem[i] = smem[r]; smem[r] = t; }
    }
    __syncthreads();
    fft_inplace(smem, 1.0f, tid, nth);

    const float Dh    = D[h];
    const float scale = 1.0f / (float)FFT_N;
    for (int l = tid; l < LEN_L; l += nth) {
        const float y = smem[l].x * scale + Dh * urow[l];
        const float s = y / (1.0f + __expf(-y));      // silu
        ybf[((size_t)b * LEN_L + l) * DIM_H + h] = (_Float16)s;
    }
}

// ---------------------------------------------------------------------------
// TDM helper: 2-D tile load Global -> LDS with row padding.
//   data_size = 2 bytes. Padding: every 16 dwords (one 32-half row) insert
//   4 dwords -> LDS row stride 80B == [..][40] halfs (bank-conflict-free and
//   16B aligned for ds_load_b128 fragment reads).
// ---------------------------------------------------------------------------
#if HAVE_TDM
__device__ __forceinline__ void tdm_load_tile(unsigned int lds_byte,
                                              unsigned long long gaddr,
                                              unsigned int tensor_d0,
                                              unsigned int tensor_d1,
                                              unsigned int tile_d0,
                                              unsigned int tile_d1,
                                              unsigned int stride0) {
    u32x4 g0 = {0u, 0u, 0u, 0u};
    g0[0] = 1u;                                   // count=1, user descriptor
    g0[1] = lds_byte;                             // lds_addr
    g0[2] = (unsigned int)gaddr;                  // global_addr[31:0]
    g0[3] = (unsigned int)(gaddr >> 32) | (2u << 30);  // addr hi | type=2

    i32x8 g1 = {0, 0, 0, 0, 0, 0, 0, 0};
    // data_size=1 (2B) | pad_enable | pad_interval=3 (16 dw) | pad_amount=3 (4 dw)
    g1[0] = (int)((1u << 16) | (1u << 20) | (3u << 22) | (3u << 25));
    g1[1] = (int)((tensor_d0 & 0xffffu) << 16);                 // td0[15:0]
    g1[2] = (int)((tensor_d0 >> 16) | ((tensor_d1 & 0xffffu) << 16));
    g1[3] = (int)((tensor_d1 >> 16) | (tile_d0 << 16));         // tile_dim0
    g1[4] = (int)(tile_d1 & 0xffffu);                           // tile_dim1
    g1[5] = (int)stride0;                                       // dim0 stride lo
    g1[6] = 0;
    g1[7] = 0;

    i32x4 g2 = {0, 0, 0, 0};
    i32x4 g3 = {0, 0, 0, 0};
#if __clang_major__ >= 23
    i32x8 g4 = {0, 0, 0, 0, 0, 0, 0, 0};
    __builtin_amdgcn_tensor_load_to_lds(g0, g1, g2, g3, g4, 0);
#else
    __builtin_amdgcn_tensor_load_to_lds(g0, g1, g2, g3, 0);
#endif
}
#endif

// ---------------------------------------------------------------------------
// Kernel 4: WMMA GEMM  z[M=16384, N=2048] = A[M,K=1024] * B[K,N]
//   Block tile 128(M) x 256(N), 8 waves (2 x 4), each wave 64x64 via a 4x4
//   grid of v_wmma_f32_16x16x32_f16 accumulators. Tiles double-buffered in
//   LDS; fed by the Tensor Data Mover (wave 0 issues, TENSORcnt-synced),
//   with a plain cooperative-load fallback when the builtin is unavailable.
// ---------------------------------------------------------------------------
__global__ __launch_bounds__(256)
void gemm_kernel(const _Float16* __restrict__ A,
                 const _Float16* __restrict__ Bw,
                 float* __restrict__ Z) {
    __shared__ _Float16 At[2][128][40];   // row stride 80B
    __shared__ _Float16 Bt[2][256][40];   // [col][k]

    const int tid  = threadIdx.x;
    const int lane = tid & 31;
    const int wave = tid >> 5;   // 0..7
    const int wy   = wave & 1;   // 2 waves along M (64 rows each)
    const int wx   = wave >> 1;  // 4 waves along N (64 cols each)
    const int m0   = blockIdx.y * 128;
    const int n0   = blockIdx.x * 256;
    const int NS   = DIM_H / 32; // 32 k-steps

    v8f acc[4][4];
    for (int i = 0; i < 4; ++i)
        for (int j = 0; j < 4; ++j)
            for (int v = 0; v < 8; ++v) acc[i][j][v] = 0.0f;

    const int arow0 = wy * 64 + (lane & 15);
    const int akb   = (lane >> 4) * 8;    // A frag: K 0-7/16-23 | 8-15/24-31
    const int bcol0 = wx * 64 + (lane & 15);
    const int bkb   = (lane >> 4) * 8;    // B frag: K 0-7/16-23 | 8-15/24-31

#if HAVE_TDM
    const unsigned long long aBase = (unsigned long long)(uintptr_t)A;
    const unsigned long long bBase = (unsigned long long)(uintptr_t)Bw;
    if (wave == 0) {
        tdm_load_tile((unsigned int)(uintptr_t)&At[0][0][0],
                      aBase + (((unsigned long long)m0 << 10) << 1),
                      DIM_H, 16384u, 32u, 128u, DIM_H);
        tdm_load_tile((unsigned int)(uintptr_t)&Bt[0][0][0],
                      bBase + (((unsigned long long)n0 << 10) << 1),
                      DIM_H, 2048u, 32u, 256u, DIM_H);
    }
#else
    {   // prologue load of buffer 0 (k0 = 0)
        const int r  = tid >> 1;
        const int kh = (tid & 1) * 16;
        const v8h* ga = (const v8h*)(A + (((size_t)(m0 + r)) << 10) + kh);
        *(v8h*)&At[0][r][kh]     = ga[0];
        *(v8h*)&At[0][r][kh + 8] = ga[1];
        for (int half = 0; half < 2; ++half) {
            const int c = tid + half * 128;
            const v8h* gb = (const v8h*)(Bw + (((size_t)(n0 + c)) << 10));
            *(v8h*)&Bt[0][c][0] = gb[0];
            *(v8h*)&Bt[0][c][8] = gb[1];
            *(v8h*)&Bt[0][c][16] = gb[2];
            *(v8h*)&Bt[0][c][24] = gb[3];
        }
    }
#endif

    for (int i = 0; i < NS; ++i) {
        const int cur = i & 1;
#if HAVE_TDM
        if (wave == 0) {
            if (i + 1 < NS) {
                const unsigned long long koff = ((unsigned long long)(i + 1) * 32ull) << 1;
                tdm_load_tile((unsigned int)(uintptr_t)&At[cur ^ 1][0][0],
                              aBase + (((unsigned long long)m0 << 10) << 1) + koff,
                              DIM_H, 16384u, 32u, 128u, DIM_H);
                tdm_load_tile((unsigned int)(uintptr_t)&Bt[cur ^ 1][0][0],
                              bBase + (((unsigned long long)n0 << 10) << 1) + koff,
                              DIM_H, 2048u, 32u, 256u, DIM_H);
                __builtin_amdgcn_s_wait_tensorcnt(2);  // first pair landed
            } else {
                __builtin_amdgcn_s_wait_tensorcnt(0);
            }
        }
#else
        if (i + 1 < NS) {
            const int k0n = (i + 1) * 32;
            const int r  = tid >> 1;
            const int kh = (tid & 1) * 16;
            const v8h* ga = (const v8h*)(A + (((size_t)(m0 + r)) << 10) + k0n + kh);
            *(v8h*)&At[cur ^ 1][r][kh]     = ga[0];
            *(v8h*)&At[cur ^ 1][r][kh + 8] = ga[1];
            for (int half = 0; half < 2; ++half) {
                const int c = tid + half * 128;
                const v8h* gb = (const v8h*)(Bw + (((size_t)(n0 + c)) << 10) + k0n);
                *(v8h*)&Bt[cur ^ 1][c][0]  = gb[0];
                *(v8h*)&Bt[cur ^ 1][c][8]  = gb[1];
                *(v8h*)&Bt[cur ^ 1][c][16] = gb[2];
                *(v8h*)&Bt[cur ^ 1][c][24] = gb[3];
            }
        }
#endif
        __syncthreads();     // buf[cur] ready for everyone

        v16h afr[4];
        for (int mt = 0; mt < 4; ++mt) {
            const v8h lo = *(const v8h*)&At[cur][arow0 + mt * 16][akb];
            const v8h hi = *(const v8h*)&At[cur][arow0 + mt * 16][akb + 16];
            for (int e = 0; e < 8; ++e) { afr[mt][e] = lo[e]; afr[mt][e + 8] = hi[e]; }
        }
        for (int nt = 0; nt < 4; ++nt) {
            v16h bf;
            {
                const v8h lo = *(const v8h*)&Bt[cur][bcol0 + nt * 16][bkb];
                const v8h hi = *(const v8h*)&Bt[cur][bcol0 + nt * 16][bkb + 16];
                for (int e = 0; e < 8; ++e) { bf[e] = lo[e]; bf[e + 8] = hi[e]; }
            }
            for (int mt = 0; mt < 4; ++mt) {
                acc[mt][nt] = __builtin_amdgcn_wmma_f32_16x16x32_f16(
                    false, afr[mt], false, bf, (short)0, acc[mt][nt], false, false);
            }
        }
        __syncthreads();     // everyone done reading buf[cur]
    }

    for (int mt = 0; mt < 4; ++mt) {
        const int Mb = m0 + wy * 64 + mt * 16 + (lane >> 4) * 8;
        for (int nt = 0; nt < 4; ++nt) {
            const int N = n0 + wx * 64 + nt * 16 + (lane & 15);
            float* zp = Z + (size_t)Mb * 2048 + N;
            for (int v = 0; v < 8; ++v) zp[(size_t)v * 2048] = acc[mt][nt][v];
        }
    }
}

// ---------------------------------------------------------------------------
// Kernel 5: GLU with 32x32 LDS transpose.
//   reads z (B*L, 2H) coalesced along h; writes out (B,H,L) coalesced along l.
// ---------------------------------------------------------------------------
__global__ __launch_bounds__(256)
void glu_kernel(const float* __restrict__ Z,
                const float* __restrict__ bias,
                float* __restrict__ out) {
    __shared__ float tile[32][33];
    const int tid = threadIdx.x;
    const int l0  = blockIdx.x * 32;
    const int h0  = blockIdx.y * 32;
    const int b   = blockIdx.z;

    {   // gather + GLU: rows = l, cols = h (coalesced 16B reads of z)
        const int lr = tid >> 3;
        const int hc = (tid & 7) * 4;
        const size_t m    = ((size_t)b << 12) + (l0 + lr);
        const size_t base = m * 2048 + (h0 + hc);
        const float4 za = *(const float4*)&Z[base];
        const float4 zg = *(const float4*)&Z[base + DIM_H];
        const float4 ba = *(const float4*)&bias[h0 + hc];
        const float4 bg = *(const float4*)&bias[DIM_H + h0 + hc];
        tile[lr][hc + 0] = (za.x + ba.x) * (1.0f / (1.0f + __expf(-(zg.x + bg.x))));
        tile[lr][hc + 1] = (za.y + ba.y) * (1.0f / (1.0f + __expf(-(zg.y + bg.y))));
        tile[lr][hc + 2] = (za.z + ba.z) * (1.0f / (1.0f + __expf(-(zg.z + bg.z))));
        tile[lr][hc + 3] = (za.w + ba.w) * (1.0f / (1.0f + __expf(-(zg.w + bg.w))));
    }
    __syncthreads();
    {   // scatter transposed: rows = h, cols = l (coalesced 16B writes of out)
        const int hr = tid >> 3;
        const int lc = (tid & 7) * 4;
        float4 o;
        o.x = tile[lc + 0][hr];
        o.y = tile[lc + 1][hr];
        o.z = tile[lc + 2][hr];
        o.w = tile[lc + 3][hr];
        *(float4*)&out[(((size_t)b << 10) + (h0 + hr)) * LEN_L + l0 + lc] = o;
    }
}

// ---------------------------------------------------------------------------
extern "C" void kernel_launch(void* const* d_in, const int* in_sizes, int n_in,
                              void* d_out, int out_size, void* d_ws, size_t ws_size,
                              hipStream_t stream) {
    (void)in_sizes; (void)n_in; (void)out_size; (void)ws_size;

    const float* u    = (const float*)d_in[0];   // (B,H,L)
    const float* kern = (const float*)d_in[1];   // (C,H,2L)
    const float* D    = (const float*)d_in[2];   // (C,H)
    const float* W    = (const float*)d_in[3];   // (2H, C*H)
    const float* bias = (const float*)d_in[4];   // (2H,)
    float* out = (float*)d_out;                  // (B,H,L)

    char* ws = (char*)d_ws;
    float2*   kf  = (float2*)(ws);                          //  64 MB
    _Float16* ybf = (_Float16*)(ws + (64ull << 20));        //  32 MB
    _Float16* wf  = (_Float16*)(ws + (96ull << 20));        //   4 MB
    float*    z   = (float*)(ws + (100ull << 20));          // 128 MB

    const size_t fft_lds = (size_t)FFT_N * sizeof(float2);  // 64 KB

    kfft_kernel <<<DIM_H,                     256, fft_lds, stream>>>(kern, kf);
    wconv_kernel<<<(2 * DIM_H * DIM_H) / 256, 256, 0,       stream>>>(W, wf);
    conv_kernel <<<BATCH_B * DIM_H,           256, fft_lds, stream>>>(u, D, kf, ybf);
    gemm_kernel <<<dim3(8, 128, 1),           256, 0,       stream>>>(ybf, wf, z);
    glu_kernel  <<<dim3(LEN_L / 32, DIM_H / 32, BATCH_B), 256, 0, stream>>>(z, bias, out);
}
